// StructureModuleV47_42374147342907
// MI455X (gfx1250) — matrix operand order repaired and checked by reading, not compiled, not run on hardware
//
#include <hip/hip_runtime.h>
#include <math.h>

typedef _Float16 h16;
typedef __attribute__((ext_vector_type(16))) _Float16 v16h;
typedef __attribute__((ext_vector_type(8)))  _Float16 v8h;
typedef __attribute__((ext_vector_type(8)))  float    v8f;

constexpr int Bc = 2, Nc = 192, Cc = 384, PDc = 128, Hc = 12, P3c = 144, HIDc = 256, Ec = 128;
constexpr float CUTC = 6.0f;
constexpr float SCALEC = 0.17677669529663687f; // 32^-0.5

// ---------------- WMMA fragment loaders (CDNA5 16x16x32 f16 layout) ----------------
// A 16x32 (row-major [M,K], K mult of 32): lane lo = M; elem e -> K = k0+8*hi+e (e<8), +16 (e>=8)
// => two contiguous 8-half (16B) vector loads.
__device__ __forceinline__ v16h frag_a(const h16* __restrict__ Arow, int k0, int hi) {
  const h16* p = Arow + k0 + 8 * hi;
  v8h lo = *(const v8h*)p;
  v8h hh = *(const v8h*)(p + 16);
  return __builtin_shufflevector(lo, hh, 0, 1, 2, 3, 4, 5, 6, 7, 8, 9, 10, 11, 12, 13, 14, 15);
}
// B given TRANSPOSED as BT[N,K] row-major: lane lo = N; elem e -> K = k0+8*hi+e (e<8), +16 (e>=8)
// => two contiguous 8-half (16B) vector loads.
__device__ __forceinline__ v16h frag_b(const h16* __restrict__ Brow, int k0, int hi) {
  const h16* p = Brow + k0 + 8 * hi;
  v8h lo = *(const v8h*)p;
  v8h hh = *(const v8h*)(p + 16);
  return __builtin_shufflevector(lo, hh, 0, 1, 2, 3, 4, 5, 6, 7, 8, 9, 10, 11, 12, 13, 14, 15);
}

// ---------------- generic batched GEMM: C[M,N] = A(f16)[M,K] @ BT(f16)[N,K]^T + bias ----------------
// Requirements (guaranteed by host staging): M,N multiples of 16; K multiple of 32; rows 16B-aligned.
__global__ void k_wmma_gemm(const h16* __restrict__ A, const h16* __restrict__ BT,
                            const float* __restrict__ bias, void* __restrict__ Cout,
                            int out_half, int M, int Ncol, int K,
                            long sA, long sB, long sC) {
  int wpb  = blockDim.x >> 5;
  int wave = threadIdx.x >> 5;
  int lane = threadIdx.x & 31;
  int lo = lane & 15, hi = lane >> 4;
  int m0 = blockIdx.x * 16;
  int n0 = (blockIdx.y * wpb + wave) * 16;
  if (n0 >= Ncol) return;                 // wave-uniform exit (EXEC all-1 for WMMA)
  const h16* Arow = A + (long)blockIdx.z * sA + (long)(m0 + lo) * K;
  const h16* Brow = BT + (long)blockIdx.z * sB + (long)(n0 + lo) * K;
  float bv = (bias != nullptr) ? bias[n0 + lo] : 0.f;
  v8f acc;
#pragma unroll
  for (int r = 0; r < 8; r++) acc[r] = bv;
  for (int k0 = 0; k0 < K; k0 += 32) {
    v16h a = frag_a(Arow, k0, hi);
    v16h b = frag_b(Brow, k0, hi);
    acc = __builtin_amdgcn_wmma_f32_16x16x32_f16(false, a, false, b, (short)0, acc, false, false);
  }
#pragma unroll
  for (int r = 0; r < 8; r++) {
    long o = (long)blockIdx.z * sC + (long)(m0 + r + 8 * hi) * Ncol + (n0 + lo);
    if (out_half) ((h16*)Cout)[o] = (h16)acc[r];
    else          ((float*)Cout)[o] = acc[r];
  }
}

// ---------------- elementwise helpers ----------------
__global__ void k_copy_f32(float* __restrict__ d, const float* __restrict__ s, long n) {
  long i = (long)blockIdx.x * blockDim.x + threadIdx.x; if (i < n) d[i] = s[i];
}
__global__ void k_zero_f32(float* __restrict__ d, long n) {
  long i = (long)blockIdx.x * blockDim.x + threadIdx.x; if (i < n) d[i] = 0.f;
}
__global__ void k_f32_to_f16(h16* __restrict__ d, const float* __restrict__ s, long n) {
  long i = (long)blockIdx.x * blockDim.x + threadIdx.x; if (i < n) d[i] = (h16)s[i];
}
__global__ void k_relu_f16(h16* __restrict__ d, const float* __restrict__ s, long n) {
  long i = (long)blockIdx.x * blockDim.x + threadIdx.x; if (i < n) d[i] = (h16)fmaxf(s[i], 0.f);
}
__global__ void k_add_f16(h16* __restrict__ d, const float* __restrict__ a, const float* __restrict__ b, long n) {
  long i = (long)blockIdx.x * blockDim.x + threadIdx.x; if (i < n) d[i] = (h16)(a[i] + b[i]);
}
// transpose-pack: dst (f16, [Ntot rows, ldD=K cols]) row block at rowoff <- src (f32 [rows=K, cols])
__global__ void k_packT(h16* __restrict__ dst, int ldD, int rowoff,
                        const float* __restrict__ src, int rows, int cols, int ldS) {
  long i = (long)blockIdx.x * blockDim.x + threadIdx.x;
  if (i >= (long)rows * cols) return;
  int r = (int)(i / cols), c = (int)(i % cols);
  dst[(long)(rowoff + c) * ldD + r] = (h16)src[(long)r * ldS + c];
}

// ---------------- LayerNorm: out = LN(a (+res)) * g + b ----------------
__global__ void k_layernorm(float* __restrict__ out, const float* __restrict__ a,
                            const float* __restrict__ res, const float* __restrict__ g,
                            const float* __restrict__ bvec, int C) {
  int row = blockIdx.x;
  const float* ar = a + (long)row * C;
  const float* rr = res ? res + (long)row * C : nullptr;
  __shared__ float red[256];
  int tid = threadIdx.x;
  float s = 0.f;
  for (int c = tid; c < C; c += blockDim.x) { float v = ar[c] + (rr ? rr[c] : 0.f); s += v; }
  red[tid] = s; __syncthreads();
  for (int st = 128; st > 0; st >>= 1) { if (tid < st) red[tid] += red[tid + st]; __syncthreads(); }
  float mean = red[0] / C; __syncthreads();
  float s2 = 0.f;
  for (int c = tid; c < C; c += blockDim.x) { float v = ar[c] + (rr ? rr[c] : 0.f) - mean; s2 += v * v; }
  red[tid] = s2; __syncthreads();
  for (int st = 128; st > 0; st >>= 1) { if (tid < st) red[tid] += red[tid + st]; __syncthreads(); }
  float rstd = rsqrtf(red[0] / C + 1e-5f);
  for (int c = tid; c < C; c += blockDim.x) {
    float v = ar[c] + (rr ? rr[c] : 0.f);
    out[(long)row * C + c] = (v - mean) * rstd * g[c] + bvec[c];
  }
}

// ---------------- IPA: build augmented attention operands + point transforms ----------------
// logits = Aaug . Baug : [q*s | q_pts*s | -0.5*s*q2 | 1 | 0..] . [k | k_pts | 1 | -0.5*s*k2 | 0..]
__global__ void k_build_attn(const float* __restrict__ proj, const float* __restrict__ rot,
                             const float* __restrict__ trans,
                             h16* __restrict__ aaug, h16* __restrict__ baugT, h16* __restrict__ vcatT) {
  int idx = blockIdx.x * blockDim.x + threadIdx.x;
  if (idx >= Bc * Hc * Nc) return;
  int n = idx % Nc;
  int h = (idx / Nc) % Hc;
  int b = idx / (Nc * Hc);
  long ip = (long)b * Nc + n;
  const float* pr = proj + ip * 1584;
  const float* R  = rot + ip * 9;
  const float* tr = trans + ip * 3;
  float qp[12], kp[12], vp[12];
  float q2 = 0.f, k2 = 0.f;
#pragma unroll
  for (int p = 0; p < 4; p++) {
    const float* lq = pr + 1152 + (h * 4 + p) * 3;
    const float* lk = pr + 1296 + (h * 4 + p) * 3;
    const float* lv = pr + 1440 + (h * 4 + p) * 3;
#pragma unroll
    for (int e = 0; e < 3; e++) {
      float vq = lq[0] * R[e] + lq[1] * R[3 + e] + lq[2] * R[6 + e] + tr[e];
      float vk = lk[0] * R[e] + lk[1] * R[3 + e] + lk[2] * R[6 + e] + tr[e];
      float vv = lv[0] * R[e] + lv[1] * R[3 + e] + lv[2] * R[6 + e] + tr[e];
      qp[p * 3 + e] = vq; kp[p * 3 + e] = vk; vp[p * 3 + e] = vv;
      q2 += vq * vq; k2 += vk * vk;
    }
  }
  // A operand (row n, K=96, contiguous)
  h16* Ar = aaug + ((long)(b * Hc + h) * Nc + n) * 96;
  for (int j = 0; j < 32; j++) Ar[j] = (h16)(pr[h * 32 + j] * SCALEC);
  for (int t = 0; t < 12; t++) Ar[32 + t] = (h16)(qp[t] * SCALEC);
  Ar[44] = (h16)(-0.5f * SCALEC * q2);
  Ar[45] = (h16)1.f;
  for (int j = 46; j < 96; j++) Ar[j] = (h16)0.f;
  // B operand transposed: baugT[(bh)*192*96 + m*96 + k] (row m contiguous)
  h16* Br = baugT + ((long)(b * Hc + h) * Nc + n) * 96;
  for (int j = 0; j < 32; j++) Br[j] = (h16)pr[384 + h * 32 + j];
  for (int t = 0; t < 12; t++) Br[32 + t] = (h16)kp[t];
  Br[44] = (h16)1.f;
  Br[45] = (h16)(-0.5f * SCALEC * k2);
  for (int j = 46; j < 96; j++) Br[j] = (h16)0.f;
  // V operand transposed: vcatT[(bh)*48*192 + j*192 + m]  (col j = v(32)|v_pts(12)|0(4))
  h16* Vc = vcatT + (long)(b * Hc + h) * 48 * 192 + n;
  for (int j = 0; j < 32; j++) Vc[(long)j * 192] = (h16)pr[768 + h * 32 + j];
  for (int t = 0; t < 12; t++) Vc[(long)(32 + t) * 192] = (h16)vp[t];
  Vc[(long)44 * 192] = Vc[(long)45 * 192] = Vc[(long)46 * 192] = Vc[(long)47 * 192] = (h16)0.f;
}

// softmax over m, adding pair bias pb[(b,n,m), h] (pb row stride 16, cols 12 valid)
__global__ void k_softmax(const float* __restrict__ logits, const float* __restrict__ pb,
                          h16* __restrict__ attn) {
  int idx = blockIdx.x;            // (b*H+h)*N + n
  int tid = threadIdx.x;
  int n = idx % Nc;
  int bh = idx / Nc;
  int h = bh % Hc;
  int b = bh / Hc;
  __shared__ float sh[256];
  float v = -1e30f;
  if (tid < Nc) v = logits[(long)idx * Nc + tid] + pb[((long)(b * Nc + n) * Nc + tid) * 16 + h];
  sh[tid] = v; __syncthreads();
  for (int st = 128; st > 0; st >>= 1) { if (tid < st) sh[tid] = fmaxf(sh[tid], sh[tid + st]); __syncthreads(); }
  float mx = sh[0]; __syncthreads();
  float ex = (tid < Nc) ? expf(v - mx) : 0.f;
  sh[tid] = ex; __syncthreads();
  for (int st = 128; st > 0; st >>= 1) { if (tid < st) sh[tid] += sh[tid + st]; __syncthreads(); }
  float inv = 1.f / sh[0];
  if (tid < Nc) attn[(long)idx * Nc + tid] = (h16)(ex * inv);
}

// gather [B,H,N,48] -> ws[B,N,C] (f32) and wp[B,N,160] (f16, cols 144.. stay zero)
__global__ void k_ipa_gather(const float* __restrict__ attnout, float* __restrict__ wsout,
                             h16* __restrict__ wp16) {
  long i = (long)blockIdx.x * blockDim.x + threadIdx.x;
  if (i >= (long)Bc * Nc * 528) return;
  int j = (int)(i % 528);
  long bn = i / 528;
  int n = (int)(bn % Nc);
  int b = (int)(bn / Nc);
  if (j < 384) {
    int h = j >> 5, c = j & 31;
    wsout[bn * 384 + j] = attnout[(((long)(b * Hc + h) * Nc) + n) * 48 + c];
  } else {
    int jj = j - 384;
    int h = jj / 12, r = jj % 12;
    wp16[bn * 160 + jj] = (h16)attnout[(((long)(b * Hc + h) * Nc) + n) * 48 + 32 + r];
  }
}

// ---------------- frame update (Rodrigues); rp row stride 16 ----------------
__global__ void k_frame_update(float* __restrict__ rot, float* __restrict__ trans,
                               const float* __restrict__ rp) {
  int i = blockIdx.x * blockDim.x + threadIdx.x;
  if (i >= Bc * Nc) return;
  const float* r = rp + (long)i * 16;
  float a0 = r[0], a1 = r[1], a2 = r[2];
  float ang = fmaxf(sqrtf(a0 * a0 + a1 * a1 + a2 * a2), 1e-8f);
  float ax = a0 / ang, ay = a1 / ang, az = a2 / ang;
  float s = sinf(ang), c = cosf(ang), o = 1.f - c;
  float R[9];
  R[0] = 1.f - o * (ay * ay + az * az); R[1] = -s * az + o * ax * ay; R[2] =  s * ay + o * ax * az;
  R[3] =  s * az + o * ax * ay;         R[4] = 1.f - o * (ax * ax + az * az); R[5] = -s * ax + o * ay * az;
  R[6] = -s * ay + o * ax * az;         R[7] =  s * ax + o * ay * az; R[8] = 1.f - o * (ax * ax + ay * ay);
  float ro[9];
#pragma unroll
  for (int j = 0; j < 9; j++) ro[j] = rot[(long)i * 9 + j];
  float t0 = r[3], t1 = r[4], t2 = r[5];          // trans update uses OLD rot
  trans[(long)i * 3 + 0] += t0 * ro[0] + t1 * ro[3] + t2 * ro[6];
  trans[(long)i * 3 + 1] += t0 * ro[1] + t1 * ro[4] + t2 * ro[7];
  trans[(long)i * 3 + 2] += t0 * ro[2] + t1 * ro[5] + t2 * ro[8];
#pragma unroll
  for (int rr = 0; rr < 3; rr++)
#pragma unroll
    for (int cc = 0; cc < 3; cc++)
      rot[(long)i * 9 + rr * 3 + cc] =
          ro[rr * 3 + 0] * R[cc] + ro[rr * 3 + 1] * R[3 + cc] + ro[rr * 3 + 2] * R[6 + cc];
}

// ---------------- EGNN: edge features (silu(dist*eW1+eb1)) ----------------
__global__ void k_edge_feat(const float* __restrict__ x, const float* __restrict__ eW1,
                            const float* __restrict__ eb1, h16* __restrict__ feat) {
  long eidx = blockIdx.x;
  int b = (int)(eidx / (Nc * Nc));
  int r = (int)(eidx % (Nc * Nc));
  int i = r / Nc, j = r % Nc;
  float dx = x[(long)(b * Nc + i) * 3 + 0] - x[(long)(b * Nc + j) * 3 + 0];
  float dy = x[(long)(b * Nc + i) * 3 + 1] - x[(long)(b * Nc + j) * 3 + 1];
  float dz = x[(long)(b * Nc + i) * 3 + 2] - x[(long)(b * Nc + j) * 3 + 2];
  float dist = sqrtf(dx * dx + dy * dy + dz * dz);
  int e = threadIdx.x;
  float p = dist * eW1[e] + eb1[e];
  feat[eidx * Ec + e] = (h16)(p / (1.f + expf(-p)));
}

// ---------------- EGNN fused per-destination reduce ----------------
// one block per (b,d): pre = ea@W1cat + h_src + h_dst(+bias); silu; adjacency mask;
// node cols -> Sum_s into nodeacc; coord cols -> w[s] = dot(silu, cW2) -> x_agg
__global__ void k_egnn_edge_reduce(const h16* __restrict__ ea, const float* __restrict__ hterm,
                                   const float* __restrict__ x, const h16* __restrict__ w1catT,
                                   const float* __restrict__ cW2, h16* __restrict__ nodeacc,
                                   float* __restrict__ deg, float* __restrict__ xout) {
  int d = blockIdx.x, b = blockIdx.y;
  int tid = threadIdx.x;
  __shared__ float xs[Nc * 3];
  __shared__ float dist_s[Nc];
  __shared__ float nacc[HIDc];
  __shared__ float wsum[Nc];
  __shared__ float r4[1024];
  for (int t = tid; t < Nc * 3; t += 256) xs[t] = x[(long)b * Nc * 3 + t];
  if (tid < HIDc) nacc[tid] = 0.f;
  if (tid < Nc) wsum[tid] = 0.f;
  __syncthreads();
  float xd0 = xs[d * 3 + 0], xd1 = xs[d * 3 + 1], xd2 = xs[d * 3 + 2];
  if (tid < Nc) {
    float dx = xs[tid * 3] - xd0, dy = xs[tid * 3 + 1] - xd1, dz = xs[tid * 3 + 2] - xd2;
    dist_s[tid] = sqrtf(dx * dx + dy * dy + dz * dz);
  }
  __syncthreads();
  int wave = tid >> 5, lane = tid & 31, lo = lane & 15, hi = lane >> 4;
  const h16* eabase = ea + (long)(b * Nc + d) * Nc * Ec;   // ea symmetric: ea[b,d,s,:]
  const float* htd = hterm + (long)(b * Nc + d) * 1024;
  for (int ct4 = 0; ct4 < 4; ct4++) {
    int ct = wave * 4 + ct4;
    int col = ct * 16 + lo;                // 0..511 : [node 256 | coord 256]
    bool isnode = col < HIDc;
    int cc = isnode ? col : col - HIDc;
    const h16* Brow = w1catT + (long)col * Ec;
    v16h b0 = frag_b(Brow, 0,  hi);
    v16h b1 = frag_b(Brow, 32, hi);
    v16h b2 = frag_b(Brow, 64, hi);
    v16h b3 = frag_b(Brow, 96, hi);
    float offv = htd[isnode ? (256 + cc) : (768 + cc)];    // h_dst term (+ bias folded in)
    float w2 = isnode ? 0.f : cW2[cc];
    float nlocal = 0.f;
    for (int st = 0; st < Nc / 16; st++) {
      int s0 = st * 16;
      v8f acc;
#pragma unroll
      for (int r = 0; r < 8; r++) acc[r] = 0.f;
      const h16* Arow = eabase + (long)(s0 + lo) * Ec;
      v16h a;
      a = frag_a(Arow, 0,  hi); acc = __builtin_amdgcn_wmma_f32_16x16x32_f16(false, a, false, b0, (short)0, acc, false, false);
      a = frag_a(Arow, 32, hi); acc = __builtin_amdgcn_wmma_f32_16x16x32_f16(false, a, false, b1, (short)0, acc, false, false);
      a = frag_a(Arow, 64, hi); acc = __builtin_amdgcn_wmma_f32_16x16x32_f16(false, a, false, b2, (short)0, acc, false, false);
      a = frag_a(Arow, 96, hi); acc = __builtin_amdgcn_wmma_f32_16x16x32_f16(false, a, false, b3, (short)0, acc, false, false);
      float wr[8];
#pragma unroll
      for (int r = 0; r < 8; r++) {
        int s = s0 + r + 8 * hi;
        float p = acc[r] + hterm[(long)(b * Nc + s) * 1024 + (isnode ? cc : (512 + cc))] + offv;
        float sv = p / (1.f + expf(-p));
        float ds = dist_s[s];
        sv = (ds < CUTC && ds > 1e-6f) ? sv : 0.f;
        nlocal += isnode ? sv : 0.f;
        wr[r] = sv * w2;
      }
      if (!isnode) {
#pragma unroll
        for (int r = 0; r < 8; r++) {
          float wv = wr[r];
          wv += __shfl_xor(wv, 1, 32);
          wv += __shfl_xor(wv, 2, 32);
          wv += __shfl_xor(wv, 4, 32);
          wv += __shfl_xor(wv, 8, 32);
          if (lo == 0) atomicAdd(&wsum[s0 + r + 8 * hi], wv);
        }
      }
    }
    if (isnode) atomicAdd(&nacc[col], nlocal);
  }
  __syncthreads();
  if (tid < HIDc) nodeacc[(long)(b * Nc + d) * HIDc + tid] = (h16)nacc[tid];
  float p0 = 0.f, p1 = 0.f, p2 = 0.f, dg = 0.f;
  for (int s = tid; s < Nc; s += 256) {
    float ds = dist_s[s];
    if (ds < CUTC && ds > 1e-6f) {
      dg += 1.f;
      float inv = 1.f / fmaxf(ds, 1e-8f);
      float w = wsum[s] * inv;
      p0 += w * (xs[s * 3 + 0] - xd0);
      p1 += w * (xs[s * 3 + 1] - xd1);
      p2 += w * (xs[s * 3 + 2] - xd2);
    }
  }
  r4[tid] = p0; r4[256 + tid] = p1; r4[512 + tid] = p2; r4[768 + tid] = dg;
  __syncthreads();
  for (int st = 128; st > 0; st >>= 1) {
    if (tid < st) {
      r4[tid] += r4[tid + st]; r4[256 + tid] += r4[256 + tid + st];
      r4[512 + tid] += r4[512 + tid + st]; r4[768 + tid] += r4[768 + tid + st];
    }
    __syncthreads();
  }
  if (tid == 0) {
    long o = (long)(b * Nc + d) * 3;
    xout[o + 0] = xd0 + r4[0];
    xout[o + 1] = xd1 + r4[256];
    xout[o + 2] = xd2 + r4[512];
    deg[b * Nc + d] = r4[768];
  }
}

// h = h + nodeacc@nW2 + deg*nb2
__global__ void k_e3_epilogue(float* __restrict__ s, const float* __restrict__ t,
                              const float* __restrict__ dg, const float* __restrict__ nb2v) {
  int i = blockIdx.x * blockDim.x + threadIdx.x;
  if (i >= Bc * Nc * Cc) return;
  int row = i / Cc, c = i % Cc;
  s[i] = s[i] + t[i] + dg[row] * nb2v[c];
}

// ---------------- host orchestration ----------------
static inline dim3 ew(long n) { return dim3((unsigned)((n + 255) / 256)); }

extern "C" void kernel_launch(void* const* d_in, const int* in_sizes, int n_in,
                              void* d_out, int out_size, void* d_ws, size_t ws_size,
                              hipStream_t stream) {
  (void)in_sizes; (void)n_in; (void)out_size; (void)ws_size;
  const float* in_single = (const float*)d_in[0];
  const float* in_pair   = (const float*)d_in[1];
  const float* in_rot    = (const float*)d_in[2];
  const float* in_trans  = (const float*)d_in[3];
  const float* qW  = (const float*)d_in[4];  const float* qb  = (const float*)d_in[5];
  const float* kW  = (const float*)d_in[6];  const float* kb  = (const float*)d_in[7];
  const float* vW  = (const float*)d_in[8];  const float* vb  = (const float*)d_in[9];
  const float* pbW = (const float*)d_in[10]; const float* pbb = (const float*)d_in[11];
  const float* qpW = (const float*)d_in[12]; const float* qpb = (const float*)d_in[13];
  const float* kpW = (const float*)d_in[14]; const float* kpb = (const float*)d_in[15];
  const float* vpW = (const float*)d_in[16]; const float* vpb = (const float*)d_in[17];
  const float* oW  = (const float*)d_in[18]; const float* ob  = (const float*)d_in[19];
  const float* poW = (const float*)d_in[20]; const float* pob = (const float*)d_in[21];
  const float* ipa_ng = (const float*)d_in[22]; const float* ipa_nb = (const float*)d_in[23];
  const float* ipan_g = (const float*)d_in[24]; const float* ipan_b = (const float*)d_in[25];
  const float* ruW = (const float*)d_in[26]; const float* rub = (const float*)d_in[27];
  const float* trW1 = (const float*)d_in[28]; const float* trb1 = (const float*)d_in[29];
  const float* trW2 = (const float*)d_in[30]; const float* trb2 = (const float*)d_in[31];
  const float* n_g = (const float*)d_in[32]; const float* n_b = (const float*)d_in[33];
  const float* eW1 = (const float*)d_in[34]; const float* eb1 = (const float*)d_in[35];
  const float* eW2 = (const float*)d_in[36]; const float* eb2 = (const float*)d_in[37];
  const float* nW1 = (const float*)d_in[38]; const float* nb1 = (const float*)d_in[39];
  const float* nW2 = (const float*)d_in[40]; const float* nb2 = (const float*)d_in[41];
  const float* cW1 = (const float*)d_in[42]; const float* cb1 = (const float*)d_in[43];
  const float* cW2 = (const float*)d_in[44];

  char* wsb = (char*)d_ws;
  size_t off = 0;
  auto alloc = [&](size_t bytes) -> char* {
    char* p = wsb + off;
    off += (bytes + 255) & ~(size_t)255;
    return p;
  };
  float* cur    = (float*)alloc(147456 * 4);
  float* rot    = (float*)alloc(3456 * 4);
  float* trans0 = (float*)alloc(1152 * 4);
  float* trans1 = (float*)alloc(1152 * 4);
  h16* s16        = (h16*)alloc(147456 * 2);
  h16* pairfeat16 = (h16*)alloc((size_t)9437184 * 2);   // pair(f16), later reused as edge feats
  h16* ea16       = (h16*)alloc((size_t)9437184 * 2);
  // transposed f16 weight packs (BT[N,K])
  h16* wcatT   = (h16*)alloc((size_t)1584 * 384 * 2);
  float* bcat  = (float*)alloc(1584 * 4);
  h16* pbWT    = (h16*)alloc((size_t)16 * 128 * 2);
  float* pbb16 = (float*)alloc(16 * 4);
  h16* poWT    = (h16*)alloc((size_t)384 * 160 * 2);
  h16* oWT     = (h16*)alloc((size_t)384 * 384 * 2);
  h16* ruWT    = (h16*)alloc((size_t)16 * 384 * 2);
  float* rub16 = (float*)alloc(16 * 4);
  h16* trW1T   = (h16*)alloc((size_t)1536 * 384 * 2);
  h16* trW2T   = (h16*)alloc((size_t)384 * 1536 * 2);
  h16* whcatT  = (h16*)alloc((size_t)1024 * 384 * 2);
  float* bhcat = (float*)alloc(1024 * 4);
  h16* w1catT  = (h16*)alloc((size_t)512 * 128 * 2);
  h16* eW2T    = (h16*)alloc((size_t)128 * 128 * 2);
  h16* nW2T    = (h16*)alloc((size_t)384 * 256 * 2);
  // activations / intermediates
  float* proj  = (float*)alloc((size_t)384 * 1584 * 4);
  float* pbout = (float*)alloc((size_t)73728 * 16 * 4);
  h16* aaug    = (h16*)alloc((size_t)442368 * 2);
  h16* baugT   = (h16*)alloc((size_t)442368 * 2);
  h16* vcatT   = (h16*)alloc((size_t)221184 * 2);
  float* logits  = (float*)alloc((size_t)884736 * 4);
  h16* attn16    = (h16*)alloc((size_t)884736 * 2);
  float* attnout = (float*)alloc((size_t)221184 * 4);
  float* wsbuf = (float*)alloc(147456 * 4);
  h16* wp16    = (h16*)alloc((size_t)384 * 160 * 2);
  float* t1    = (float*)alloc(147456 * 4);
  h16* t2_16   = (h16*)alloc(147456 * 2);
  float* ipaout = (float*)alloc(147456 * 4);
  float* rp     = (float*)alloc((size_t)384 * 16 * 4);
  float* hterm  = (float*)alloc(393216 * 4);
  h16* nodeacc16 = (h16*)alloc(98304 * 2);
  float* deg    = (float*)alloc(384 * 4);
  float* e3tmp  = (float*)alloc(147456 * 4);
  float* ffn1   = (float*)alloc(589824 * 4);
  h16* relu16   = (h16*)alloc(589824 * 2);
  float* ffn2   = (float*)alloc(147456 * 4);

  auto gemm = [&](const h16* A, const h16* BT, const float* bias, void* Cp, int oh,
                  int M, int Ncol, int K, int batch, long sA, long sB, long sC) {
    int wpb = (Ncol >= 64) ? 4 : ((Ncol >= 32) ? 2 : 1);
    dim3 grid((unsigned)((M + 15) / 16), (unsigned)((Ncol + 16 * wpb - 1) / (16 * wpb)), (unsigned)batch);
    dim3 blk((unsigned)(32 * wpb));
    k_wmma_gemm<<<grid, blk, 0, stream>>>(A, BT, bias, Cp, oh, M, Ncol, K, sA, sB, sC);
  };

  // ---- setup: state copies, zero-init of padded buffers, f16 transposed weight packs ----
  k_copy_f32<<<ew(147456), 256, 0, stream>>>(cur, in_single, 147456);
  k_copy_f32<<<ew(3456), 256, 0, stream>>>(rot, in_rot, 3456);
  k_copy_f32<<<ew(1152), 256, 0, stream>>>(trans0, in_trans, 1152);
  k_f32_to_f16<<<ew(9437184L), 256, 0, stream>>>(pairfeat16, in_pair, 9437184L);
  // zero padded staging (f16 buffers zeroed through f32 view; all 256B-aligned, even f16 counts)
  k_zero_f32<<<ew(1024), 256, 0, stream>>>((float*)pbWT, 1024);
  k_zero_f32<<<ew(16), 256, 0, stream>>>(pbb16, 16);
  k_zero_f32<<<ew(30720), 256, 0, stream>>>((float*)poWT, 30720);
  k_zero_f32<<<ew(3072), 256, 0, stream>>>((float*)ruWT, 3072);
  k_zero_f32<<<ew(16), 256, 0, stream>>>(rub16, 16);
  k_zero_f32<<<ew(30720), 256, 0, stream>>>((float*)wp16, 30720);
  k_zero_f32<<<ew(1024), 256, 0, stream>>>(bhcat, 1024);
  // fused projection weights: wcatT rows = [q|k|v|qp|kp|vp] outputs, K=384
  k_packT<<<ew(147456), 256, 0, stream>>>(wcatT, 384, 0,    qW, 384, 384, 384);
  k_packT<<<ew(147456), 256, 0, stream>>>(wcatT, 384, 384,  kW, 384, 384, 384);
  k_packT<<<ew(147456), 256, 0, stream>>>(wcatT, 384, 768,  vW, 384, 384, 384);
  k_packT<<<ew(55296), 256, 0, stream>>>(wcatT, 384, 1152, qpW, 384, 144, 144);
  k_packT<<<ew(55296), 256, 0, stream>>>(wcatT, 384, 1296, kpW, 384, 144, 144);
  k_packT<<<ew(55296), 256, 0, stream>>>(wcatT, 384, 1440, vpW, 384, 144, 144);
  k_copy_f32<<<ew(384), 256, 0, stream>>>(bcat + 0,    qb, 384);
  k_copy_f32<<<ew(384), 256, 0, stream>>>(bcat + 384,  kb, 384);
  k_copy_f32<<<ew(384), 256, 0, stream>>>(bcat + 768,  vb, 384);
  k_copy_f32<<<ew(144), 256, 0, stream>>>(bcat + 1152, qpb, 144);
  k_copy_f32<<<ew(144), 256, 0, stream>>>(bcat + 1296, kpb, 144);
  k_copy_f32<<<ew(144), 256, 0, stream>>>(bcat + 1440, vpb, 144);
  // other transposed packs
  k_packT<<<ew(1536), 256, 0, stream>>>(pbWT, 128, 0, pbW, 128, 12, 12);
  k_copy_f32<<<ew(12), 256, 0, stream>>>(pbb16, pbb, 12);
  k_packT<<<ew(55296), 256, 0, stream>>>(poWT, 160, 0, poW, 144, 384, 384);
  k_packT<<<ew(147456), 256, 0, stream>>>(oWT, 384, 0, oW, 384, 384, 384);
  k_packT<<<ew(2304), 256, 0, stream>>>(ruWT, 384, 0, ruW, 384, 6, 6);
  k_copy_f32<<<ew(6), 256, 0, stream>>>(rub16, rub, 6);
  k_packT<<<ew(589824), 256, 0, stream>>>(trW1T, 384, 0, trW1, 384, 1536, 1536);
  k_packT<<<ew(589824), 256, 0, stream>>>(trW2T, 1536, 0, trW2, 1536, 384, 384);
  // h-term weights: whcatT rows = [nW1_src|nW1_dst|cW1_src|cW1_dst], K=384; bias on dst halves
  k_packT<<<ew(98304), 256, 0, stream>>>(whcatT, 384, 0,   nW1, 384, 256, 256);
  k_packT<<<ew(98304), 256, 0, stream>>>(whcatT, 384, 256, nW1 + (size_t)384 * 256, 384, 256, 256);
  k_packT<<<ew(98304), 256, 0, stream>>>(whcatT, 384, 512, cW1, 384, 256, 256);
  k_packT<<<ew(98304), 256, 0, stream>>>(whcatT, 384, 768, cW1 + (size_t)384 * 256, 384, 256, 256);
  k_copy_f32<<<ew(256), 256, 0, stream>>>(bhcat + 256, nb1, 256);
  k_copy_f32<<<ew(256), 256, 0, stream>>>(bhcat + 768, cb1, 256);
  // edge-attr weights: w1catT rows = [nW1_ea | cW1_ea], K=128
  k_packT<<<ew(32768), 256, 0, stream>>>(w1catT, 128, 0,   nW1 + (size_t)768 * 256, 128, 256, 256);
  k_packT<<<ew(32768), 256, 0, stream>>>(w1catT, 128, 256, cW1 + (size_t)768 * 256, 128, 256, 256);
  k_packT<<<ew(16384), 256, 0, stream>>>(eW2T, 128, 0, eW2, 128, 128, 128);
  k_packT<<<ew(98304), 256, 0, stream>>>(nW2T, 256, 0, nW2, 256, 384, 384);
  // pair bias (loop-invariant): [B*N*N,128] @ [128,16pad] -> pbout stride 16
  gemm(pairfeat16, pbWT, pbb16, pbout, 0, 73728, 16, 128, 1, 0, 0, 0);

  for (int bi = 0; bi < 2; bi++) {
    float* xin  = (bi == 0) ? trans0 : trans1;
    float* xout = (bi == 0) ? trans1 : trans0;
    // ---- IPA ----
    k_f32_to_f16<<<ew(147456), 256, 0, stream>>>(s16, cur, 147456);
    gemm(s16, wcatT, bcat, proj, 0, 384, 1584, 384, 1, 0, 0, 0);
    k_build_attn<<<ew(4608), 256, 0, stream>>>(proj, rot, xin, aaug, baugT, vcatT);
    gemm(aaug, baugT, nullptr, logits, 0, 192, 192, 96, 24, 192L * 96, 192L * 96, 192L * 192);
    k_softmax<<<dim3(4608), dim3(256), 0, stream>>>(logits, pbout, attn16);
    gemm(attn16, vcatT, nullptr, attnout, 0, 192, 48, 192, 24, 192L * 192, 48L * 192, 192L * 48);
    k_ipa_gather<<<ew((long)Bc * Nc * 528), 256, 0, stream>>>(attnout, wsbuf, wp16);
    gemm(wp16, poWT, pob, t1, 0, 384, 384, 160, 1, 0, 0, 0);
    k_add_f16<<<ew(147456), 256, 0, stream>>>(t2_16, t1, wsbuf, 147456);
    gemm(t2_16, oWT, ob, ipaout, 0, 384, 384, 384, 1, 0, 0, 0);
    k_layernorm<<<dim3(384), dim3(256), 0, stream>>>(cur, cur, ipaout, ipa_ng, ipa_nb, 384);
    k_layernorm<<<dim3(384), dim3(256), 0, stream>>>(cur, cur, nullptr, ipan_g, ipan_b, 384);
    // ---- frame update ----
    k_f32_to_f16<<<ew(147456), 256, 0, stream>>>(s16, cur, 147456);
    gemm(s16, ruWT, rub16, rp, 0, 384, 16, 384, 1, 0, 0, 0);
    k_frame_update<<<ew(384), 256, 0, stream>>>(rot, xin, rp);
    // ---- EGNN ----
    gemm(s16, whcatT, bhcat, hterm, 0, 384, 1024, 384, 1, 0, 0, 0);
    k_edge_feat<<<dim3(73728), dim3(128), 0, stream>>>(xin, eW1, eb1, pairfeat16);
    gemm(pairfeat16, eW2T, eb2, ea16, 1, 73728, 128, 128, 1, 0, 0, 0);
    k_egnn_edge_reduce<<<dim3(192, 2), dim3(256), 0, stream>>>(ea16, hterm, xin, w1catT, cW2,
                                                               nodeacc16, deg, xout);
    gemm(nodeacc16, nW2T, nullptr, e3tmp, 0, 384, 384, 256, 1, 0, 0, 0);
    k_e3_epilogue<<<ew(147456), 256, 0, stream>>>(cur, e3tmp, deg, nb2);
    // ---- transition ----
    k_f32_to_f16<<<ew(147456), 256, 0, stream>>>(s16, cur, 147456);
    gemm(s16, trW1T, trb1, ffn1, 0, 384, 1536, 384, 1, 0, 0, 0);
    k_relu_f16<<<ew(589824), 256, 0, stream>>>(relu16, ffn1, 589824);
    gemm(relu16, trW2T, trb2, ffn2, 0, 384, 384, 1536, 1, 0, 0, 0);
    k_layernorm<<<dim3(384), dim3(256), 0, stream>>>(cur, cur, ffn2, n_g, n_b, 384);
  }
  float* outp = (float*)d_out;
  k_copy_f32<<<ew(147456), 256, 0, stream>>>(outp, cur, 147456);
  k_copy_f32<<<ew(1152), 256, 0, stream>>>(outp + 147456, trans0, 1152);  // block0->trans1, block1->trans0
}
